// Network_49469433315423
// MI455X (gfx1250) — matrix operand order repaired and checked by reading, not compiled, not run on hardware
//
#include <hip/hip_runtime.h>

// LSTM stack (10 layers, D=25, B=512, T=2048) + FC(25->7) + softmax.
// Latency-bound recurrence -> minimize dependent chain per step:
//   bf16 WMMA 16x16x32 (2 chained WMMAs per 16-gate tile, fp32 accum),
//   all weights LDS-resident (CDNA5: 320KB/WGP), fp32 cell state,
//   FC/softmax + x-prefetch hidden on the spare wave.
//   Activations: branchless sigmoid/tanh via v_exp_f32 + v_rcp_f32.

#define DIM      25
#define LAYERS   10
#define NCLS     7
#define T_STEPS  2048
#define BATCH    512
#define BTILE    16

// LDS element strides (chosen for 16B alignment of b128 DS loads + bank spread)
#define WSTR 88    // bf16 per weight row (K padded 50 -> 64)
#define ASTR 88    // bf16 per activation row
#define GSTR 116   // f32 per gate-buffer row

typedef __attribute__((ext_vector_type(16))) __bf16 v16bf;
typedef __attribute__((ext_vector_type(8)))  __bf16 v8bf;
typedef __attribute__((ext_vector_type(8)))  float  v8f;

__device__ __forceinline__ float fast_rcp(float x) {
    return __builtin_amdgcn_rcpf(x);          // v_rcp_f32 (~1 ulp)
}
__device__ __forceinline__ float fast_sigmoid(float x) {
    return fast_rcp(1.0f + __expf(-x));       // v_exp_f32 + v_rcp_f32
}
__device__ __forceinline__ float fast_tanh(float x) {
    // tanh(x) = 2*sigmoid(2x) - 1 ; saturates correctly for |x| large
    return __builtin_fmaf(2.0f, fast_sigmoid(2.0f * x), -1.0f);
}

__device__ __forceinline__ void fc_softmax(int trow, int b0, int lane,
                                           const float* __restrict__ H9,
                                           const float* __restrict__ FW,
                                           const float* __restrict__ FB,
                                           float* __restrict__ out) {
    if (lane >= 16) return;
    float h[DIM];
#pragma unroll
    for (int d = 0; d < DIM; ++d) h[d] = H9[lane * DIM + d];
    float lg[NCLS];
    float mx = -1e30f;
#pragma unroll
    for (int j = 0; j < NCLS; ++j) {
        float s = FB[j];
#pragma unroll
        for (int d = 0; d < DIM; ++d) s += FW[j * DIM + d] * h[d];
        lg[j] = s;
        mx = fmaxf(mx, s);
    }
    float den = 0.0f;
#pragma unroll
    for (int j = 0; j < NCLS; ++j) { lg[j] = __expf(lg[j] - mx); den += lg[j]; }
    float inv = fast_rcp(den);
    size_t base = ((size_t)trow * BATCH + b0 + lane) * NCLS;
#pragma unroll
    for (int j = 0; j < NCLS; ++j) out[base + j] = lg[j] * inv;
}

__global__ __launch_bounds__(256, 1)
void lstm_stack_kernel(const float* __restrict__ x,    // [T,B,25]
                       const float* __restrict__ h0,   // [L,B,25]
                       const float* __restrict__ c0,   // [L,B,25]
                       const float* __restrict__ Wih,  // [L,100,25]
                       const float* __restrict__ Whh,  // [L,100,25]
                       const float* __restrict__ bias, // [L,100]
                       const float* __restrict__ fcw,  // [7,25]
                       const float* __restrict__ fcb,  // [7]
                       float* __restrict__ out)        // [T*B,7]
{
    extern __shared__ char smem[];
    __bf16* Wl = (__bf16*)(smem);                 // [L*112][WSTR] bf16   197120 B
    __bf16* Ab = (__bf16*)(smem + 197120);        // [L*16][ASTR] bf16    28160 B
    float*  Cs = (float*)(smem + 225280);         // [L*16*25] f32        16000 B
    float*  G  = (float*)(smem + 241280);         // [16][GSTR] f32        7424 B
    float*  Bl = (float*)(smem + 248704);         // [L*112] f32           4480 B
    float*  H9 = (float*)(smem + 253184);         // [16*25] f32           1600 B
    float*  FW = (float*)(smem + 254784);         // [7*25] f32             704 B
    float*  FB = (float*)(smem + 255488);         // [7] f32

    const int tid   = threadIdx.x;
    const int lane  = tid & 31;
    const int wv    = tid >> 5;
    const int m     = lane & 15;   // A: batch row / B: gate col within tile
    const int hq    = lane >> 4;   // lane half
    const int b0    = blockIdx.x * BTILE;

    // ---------------- init: stage weights/states into LDS ----------------
    // Concat weights [Wih | Whh | 0-pad] -> bf16, rows padded 100->112
    for (int idx = tid; idx < LAYERS * 112 * 64; idx += 256) {
        int l  = idx / (112 * 64);
        int r2 = idx - l * (112 * 64);
        int n  = r2 >> 6;
        int k  = r2 & 63;
        float v = 0.0f;
        if (n < 100) {
            if (k < 25)      v = Wih[(l * 100 + n) * 25 + k];
            else if (k < 50) v = Whh[(l * 100 + n) * 25 + (k - 25)];
        }
        Wl[(l * 112 + n) * WSTR + k] = (__bf16)v;
    }
    for (int idx = tid; idx < LAYERS * 112; idx += 256) {
        int l = idx / 112, n = idx - l * 112;
        Bl[idx] = (n < 100) ? bias[l * 100 + n] : 0.0f;
    }
    // h state part of A (cols 25..49) from h0
    for (int idx = tid; idx < LAYERS * 400; idx += 256) {
        int l = idx / 400, e = idx - l * 400;
        int r = e / 25, d = e - r * 25;
        Ab[(l * 16 + r) * ASTR + 25 + d] =
            (__bf16)h0[((size_t)l * BATCH + b0 + r) * DIM + d];
    }
    // zero K-pad cols 50..63 (avoid NaN*0 from uninitialized LDS)
    for (int idx = tid; idx < LAYERS * 16 * 14; idx += 256) {
        int l = idx / (16 * 14), e = idx - l * (16 * 14);
        int r = e / 14, j = e - r * 14;
        Ab[(l * 16 + r) * ASTR + 50 + j] = (__bf16)0.0f;
    }
    // x(0) into layer-0 input slot (cols 0..24)
    for (int idx = tid; idx < 400; idx += 256) {
        int r = idx / 25, d = idx - r * 25;
        Ab[r * ASTR + d] = (__bf16)x[((size_t)0 * BATCH + b0 + r) * DIM + d];
    }
    // cell state
    for (int idx = tid; idx < LAYERS * 400; idx += 256) {
        int l = idx / 400, e = idx - l * 400;
        int r = e / 25, d = e - r * 25;
        Cs[(l * 16 + r) * 25 + d] = c0[((size_t)l * BATCH + b0 + r) * DIM + d];
    }
    for (int idx = tid; idx < NCLS * DIM; idx += 256) FW[idx] = fcw[idx];
    if (tid < NCLS) FB[tid] = fcb[tid];

    // wave 7 register prefetch of x(1)
    float xr[13];
    if (wv == 7) {
#pragma unroll
        for (int j = 0; j < 13; ++j) {
            int e = lane + 32 * j;
            if (e < 400) {
                int r = e / 25, d = e - r * 25;
                xr[j] = x[((size_t)1 * BATCH + b0 + r) * DIM + d];
            }
        }
    }
    __syncthreads();

    // Lane-constant activation selectors (branchless: tanh(x)=2*sigmoid(2x)-1)
    const int  ng  = wv * 16 + m;                    // this lane's gate column
    const bool isg = (ng >= 50) && (ng < 75);        // g-gate block -> tanh
    const float sA = isg ? 2.0f : 1.0f;              // input scale
    const float sB = isg ? 2.0f : 1.0f;              // output scale
    const float sC = isg ? -1.0f : 0.0f;             // output offset

    // ---------------- main recurrence ----------------
    for (int t = 0; t <= T_STEPS; ++t) {
        if (t == T_STEPS) {                 // flush FC for the last step
            if (wv == 7) fc_softmax(t - 1, b0, lane, H9, FW, FB, out);
            break;
        }
        for (int l = 0; l < LAYERS; ++l) {
            if (wv < 7) {
                // ---- gate GEMM: D[16b x 16n] += A[16x64] * B[64x16] (bf16, f32 acc)
                const __bf16* arow = Ab + (l * 16 + m) * ASTR + 8 * hq;
                const __bf16* brow = Wl + (l * 112 + ng) * WSTR + 8 * hq;
                v8bf a00 = *(const v8bf*)(arow);        // K  8hq..8hq+7
                v8bf a01 = *(const v8bf*)(arow + 16);   // K 16+8hq..
                v8bf a10 = *(const v8bf*)(arow + 32);   // K 32+8hq..
                v8bf a11 = *(const v8bf*)(arow + 48);   // K 48+8hq..
                v8bf b00 = *(const v8bf*)(brow);        // K  8hq..8hq+7
                v8bf b01 = *(const v8bf*)(brow + 16);
                v8bf b10 = *(const v8bf*)(brow + 32);   // K 32+8hq..
                v8bf b11 = *(const v8bf*)(brow + 48);
                v16bf a0 = __builtin_shufflevector(a00, a01, 0,1,2,3,4,5,6,7,8,9,10,11,12,13,14,15);
                v16bf a1 = __builtin_shufflevector(a10, a11, 0,1,2,3,4,5,6,7,8,9,10,11,12,13,14,15);
                v16bf bv0 = __builtin_shufflevector(b00, b01, 0,1,2,3,4,5,6,7,8,9,10,11,12,13,14,15);
                v16bf bv1 = __builtin_shufflevector(b10, b11, 0,1,2,3,4,5,6,7,8,9,10,11,12,13,14,15);
                v8f acc = {0.f,0.f,0.f,0.f,0.f,0.f,0.f,0.f};
                acc = __builtin_amdgcn_wmma_f32_16x16x32_bf16(
                        false, a0, false, bv0, (short)0, acc, false, false);
                acc = __builtin_amdgcn_wmma_f32_16x16x32_bf16(
                        false, a1, false, bv1, (short)0, acc, false, false);
                // ---- bias + branchless gate nonlinearity, scatter to gate buffer
                if (ng < 100) {
                    float bia = Bl[l * 112 + ng];
#pragma unroll
                    for (int r = 0; r < 8; ++r) {
                        float pre = acc[r] + bia;
                        float sg  = fast_rcp(1.0f + __expf(-sA * pre));
                        G[(r + 8 * hq) * GSTR + ng] = __builtin_fmaf(sB, sg, sC);
                    }
                }
            } else {
                // ---- spare wave: FC/softmax of previous step + x prefetch
                if (l == 0 && t > 0)
                    fc_softmax(t - 1, b0, lane, H9, FW, FB, out);
                if (l == 1) {
                    if (t + 1 < T_STEPS) {              // publish x(t+1) into A0
#pragma unroll
                        for (int j = 0; j < 13; ++j) {
                            int e = lane + 32 * j;
                            if (e < 400) {
                                int r = e / 25, d = e - r * 25;
                                Ab[r * ASTR + d] = (__bf16)xr[j];
                            }
                        }
                    }
                    if (t + 2 < T_STEPS) {              // fetch x(t+2)
#pragma unroll
                        for (int j = 0; j < 13; ++j) {
                            int e = lane + 32 * j;
                            if (e < 400) {
                                int r = e / 25, d = e - r * 25;
                                xr[j] = x[((size_t)(t + 2) * BATCH + b0 + r) * DIM + d];
                            }
                        }
                    }
                }
            }
            __syncthreads();

            // ---- elementwise cell/hidden update (400 elems, all 256 threads)
#pragma unroll
            for (int pass = 0; pass < 2; ++pass) {
                int e = tid + pass * 256;
                if (e < 400) {
                    int b = e / 25, d = e - b * 25;
                    float gi = G[b * GSTR + d];
                    float gf = G[b * GSTR + 25 + d];
                    float gg = G[b * GSTR + 50 + d];
                    float go = G[b * GSTR + 75 + d];
                    float c  = __builtin_fmaf(gf, Cs[(l * 16 + b) * 25 + d], gi * gg);
                    Cs[(l * 16 + b) * 25 + d] = c;
                    float h = go * fast_tanh(c);
                    __bf16 hb = (__bf16)h;
                    Ab[(l * 16 + b) * ASTR + 25 + d] = hb;        // own h slot (t)
                    if (l < LAYERS - 1)
                        Ab[((l + 1) * 16 + b) * ASTR + d] = hb;   // next layer input
                    else
                        H9[b * 25 + d] = h;                       // top-layer -> FC
                }
            }
            __syncthreads();
        }
    }
}

extern "C" void kernel_launch(void* const* d_in, const int* in_sizes, int n_in,
                              void* d_out, int out_size, void* d_ws, size_t ws_size,
                              hipStream_t stream) {
    (void)in_sizes; (void)n_in; (void)d_ws; (void)ws_size; (void)out_size;
    const float* x   = (const float*)d_in[0];
    const float* h0  = (const float*)d_in[1];
    const float* c0  = (const float*)d_in[2];
    const float* Wih = (const float*)d_in[3];
    const float* Whh = (const float*)d_in[4];
    const float* b   = (const float*)d_in[5];
    const float* fcw = (const float*)d_in[6];
    const float* fcb = (const float*)d_in[7];
    float* out = (float*)d_out;

    const size_t shmem = 255520;   // ~250 KB LDS (weights resident; < 320 KB/WGP)
    lstm_stack_kernel<<<dim3(BATCH / BTILE), dim3(256), shmem, stream>>>(
        x, h0, c0, Wih, Whh, b, fcw, fcb, out);
}
